// MResConv_65919158059199
// MI455X (gfx1250) — matrix operand order repaired and checked by reading, not compiled, not run on hardware
//
#include <hip/hip_runtime.h>
#include <hip/hip_bf16.h>

// ---------------- problem constants (match reference setup_inputs) ----------
constexpr int kB = 2;
constexpr int kC = 128;      // channels (M of GEMM)
constexpr int kE = 65536;    // edges    (N of GEMM)
constexpr int kK = kC * 5;   // 640      (K of GEMM)
constexpr float kEps = 1e-5f;

constexpr int NT   = 128;    // edge tile per workgroup
constexpr int CB   = 32;     // channel block (one WMMA K-chunk per s-plane)
constexpr int THREADS = 256; // 8 waves (wave32)

typedef __attribute__((ext_vector_type(16))) __bf16 v16bf;
typedef __attribute__((ext_vector_type(8)))  float  v8f;

union ABu { v16bf v; uint4 u[2]; };

__device__ __forceinline__ unsigned short f2bf(float x) {
    unsigned int u = __float_as_uint(x);
    unsigned int r = (u + 0x7FFFu + ((u >> 16) & 1u)) >> 16;  // RNE
    return (unsigned short)r;
}

// ---------------- weight repack: [O,C,5] f32 -> [M=128][K=640] bf16, k=s*128+c
__global__ __launch_bounds__(256)
void prep_w_kernel(const float* __restrict__ w, unsigned short* __restrict__ Wt) {
    int i = blockIdx.x * 256 + threadIdx.x;           // over 128*128*5
    if (i >= kC * kC * 5) return;
    int s = i % 5;
    int c = (i / 5) % kC;
    int m = i / (5 * kC);
    Wt[(size_t)m * kK + s * kC + c] = f2bf(w[i]);
}

// ---------------- fused gather + bf16 WMMA GEMM: Y[b,o,e] = meshconv + bias --
__global__ __launch_bounds__(THREADS)
void conv_kernel(const float* __restrict__ X, const int* __restrict__ gemm,
                 const unsigned short* __restrict__ Wt,
                 const float* __restrict__ bias, float* __restrict__ Y) {
    __shared__ int gidx[NT][4];
    __shared__ __align__(16) unsigned short plane[5][NT][CB];  // 40 KB

    const int b    = blockIdx.y;
    const int n0   = blockIdx.x * NT;
    const int tid  = threadIdx.x;
    const int lane = tid & 31;
    const int wave = tid >> 5;
    const int m0   = wave * 16;             // 16-channel output strip per wave
    const int half = lane >> 4;             // lane half selects K sub-range

    const float* xb = X + (size_t)b * kC * kE;
    const int*   gb = gemm + ((size_t)b * kE + n0) * 4;

    // stage neighbor indices for this edge tile
    for (int i = tid; i < NT * 4; i += THREADS)
        ((int*)gidx)[i] = gb[i];

    v8f acc[8] = {};   // 8 N-tiles of 16x16 f32

    for (int cb = 0; cb < kC; cb += CB) {
        __syncthreads();  // covers gidx staging + plane reuse across blocks

        // build the 5 symmetric-feature planes in bf16: plane[s][e][c_local]
        for (int i = tid; i < CB * NT; i += THREADS) {
            int e  = i & (NT - 1);
            int cl = i >> 7;            // i / NT
            const float* xc = xb + (size_t)(cb + cl) * kE;
            float x0 = xc[n0 + e];
            float f1 = xc[gidx[e][0]];
            float f2 = xc[gidx[e][1]];
            float f3 = xc[gidx[e][2]];
            float f4 = xc[gidx[e][3]];
            plane[0][e][cl] = f2bf(x0);
            plane[1][e][cl] = f2bf(f1 + f3);
            plane[2][e][cl] = f2bf(f2 + f4);
            plane[3][e][cl] = f2bf(fabsf(f1 - f3));
            plane[4][e][cl] = f2bf(fabsf(f2 - f4));
        }
        __syncthreads();

        #pragma unroll
        for (int s = 0; s < 5; ++s) {
            // A fragment (weights): lanes 0-15 take K 0-7 & 16-23, lanes 16-31 take 8-15 & 24-31
            const unsigned short* arow =
                Wt + (size_t)(m0 + (lane & 15)) * kK + s * kC + cb;
            ABu a;
            a.u[0] = *(const uint4*)(arow + half * 8);
            a.u[1] = *(const uint4*)(arow + half * 8 + 16);

            #pragma unroll
            for (int nt = 0; nt < 8; ++nt) {
                // B fragment: lane n = nt*16 + lane%16, lanes 0-15 take K 0-7 & 16-23, lanes 16-31 take 8-15 & 24-31
                const unsigned short* bp =
                    &plane[s][nt * 16 + (lane & 15)][half * 8];
                ABu bb;
                bb.u[0] = *(const uint4*)(bp);
                bb.u[1] = *(const uint4*)(bp + 16);
                acc[nt] = __builtin_amdgcn_wmma_f32_16x16x32_bf16(
                    false, a.v, false, bb.v, (short)0, acc[nt], false, false);
            }
        }
    }

    // C/D layout: lanes 0-15 -> M = v, lanes 16-31 -> M = v + 8; N = lane % 16
    float bch[8];
    #pragma unroll
    for (int v = 0; v < 8; ++v) bch[v] = bias[m0 + v + half * 8];

    float* yb = Y + (size_t)b * kC * kE;
    #pragma unroll
    for (int nt = 0; nt < 8; ++nt) {
        int e = n0 + nt * 16 + (lane & 15);
        #pragma unroll
        for (int v = 0; v < 8; ++v) {
            int c = m0 + v + half * 8;
            yb[(size_t)c * kE + e] = acc[nt][v] + bch[v];
        }
    }
}

// ---------------- per-(b,c) mean / rstd over E ------------------------------
__global__ __launch_bounds__(256)
void stats_kernel(const float* __restrict__ Y, float* __restrict__ stats) {
    const int bc = blockIdx.x;                // 0 .. B*C-1
    const float* p = Y + (size_t)bc * kE;
    float s = 0.f, s2 = 0.f;
    for (int i = threadIdx.x * 4; i < kE; i += 256 * 4) {
        float4 v = *(const float4*)(p + i);
        s  += v.x + v.y + v.z + v.w;
        s2 += v.x * v.x + v.y * v.y + v.z * v.z + v.w * v.w;
    }
    __shared__ float rs[256], rq[256];
    rs[threadIdx.x] = s; rq[threadIdx.x] = s2;
    __syncthreads();
    for (int off = 128; off > 0; off >>= 1) {
        if (threadIdx.x < off) {
            rs[threadIdx.x] += rs[threadIdx.x + off];
            rq[threadIdx.x] += rq[threadIdx.x + off];
        }
        __syncthreads();
    }
    if (threadIdx.x == 0) {
        float mean = rs[0] * (1.f / kE);
        float var  = rq[0] * (1.f / kE) - mean * mean;
        stats[bc]            = mean;
        stats[kB * kC + bc]  = rsqrtf(var + kEps);
    }
}

// ---------------- elementwise: relu(instnorm(Y)) ----------------------------
__global__ __launch_bounds__(256)
void norm_relu_kernel(const float* __restrict__ Y, const float* __restrict__ stats,
                      float* __restrict__ Out) {
    int i = blockIdx.x * 256 + threadIdx.x;   // float4 groups, B*C*E/4 total
    int bc = i / (kE / 4);
    float m = stats[bc], r = stats[kB * kC + bc];
    float4 v = ((const float4*)Y)[i];
    v.x = fmaxf((v.x - m) * r, 0.f);
    v.y = fmaxf((v.y - m) * r, 0.f);
    v.z = fmaxf((v.z - m) * r, 0.f);
    v.w = fmaxf((v.w - m) * r, 0.f);
    ((float4*)Out)[i] = v;
}

// ---------------- elementwise: relu(instnorm(Y) + identity) -----------------
__global__ __launch_bounds__(256)
void norm_add_relu_kernel(const float* __restrict__ Y, const float* __restrict__ stats,
                          const float* __restrict__ Id, float* __restrict__ Out) {
    int i = blockIdx.x * 256 + threadIdx.x;
    int bc = i / (kE / 4);
    float m = stats[bc], r = stats[kB * kC + bc];
    float4 v = ((const float4*)Y)[i];
    float4 d = ((const float4*)Id)[i];
    v.x = fmaxf((v.x - m) * r + d.x, 0.f);
    v.y = fmaxf((v.y - m) * r + d.y, 0.f);
    v.z = fmaxf((v.z - m) * r + d.z, 0.f);
    v.w = fmaxf((v.w - m) * r + d.w, 0.f);
    ((float4*)Out)[i] = v;
}

// ---------------- driver ----------------------------------------------------
extern "C" void kernel_launch(void* const* d_in, const int* in_sizes, int n_in,
                              void* d_out, int out_size, void* d_ws, size_t ws_size,
                              hipStream_t stream) {
    const float* x    = (const float*)d_in[0];
    const int*   gemm = (const int*)  d_in[1];
    const float* w1   = (const float*)d_in[2];
    const float* b1   = (const float*)d_in[3];
    const float* w2   = (const float*)d_in[4];
    const float* b2   = (const float*)d_in[5];
    float* out = (float*)d_out;

    char* ws = (char*)d_ws;
    const size_t wtBytes  = (size_t)kC * kK * sizeof(unsigned short); // 160 KB
    unsigned short* Wt1 = (unsigned short*)ws;
    unsigned short* Wt2 = (unsigned short*)(ws + wtBytes);
    float* stats = (float*)(ws + 2 * wtBytes);
    size_t off = 2 * wtBytes + 4096;            // stats + pad, keep 256B align
    float* Y = (float*)(ws + off);              // 64 MB
    float* T = (float*)(ws + off + (size_t)kB * kC * kE * sizeof(float));

    const int elem4 = kB * kC * kE / 4;         // float4 groups
    dim3 convGrid(kE / NT, kB);

    prep_w_kernel<<<(kC * kC * 5 + 255) / 256, 256, 0, stream>>>(w1, Wt1);
    prep_w_kernel<<<(kC * kC * 5 + 255) / 256, 256, 0, stream>>>(w2, Wt2);

    const float* P = x;                          // block input / identity
    for (int blk = 0; blk < 3; ++blk) {
        conv_kernel<<<convGrid, THREADS, 0, stream>>>(P, gemm, Wt1, b1, Y);
        stats_kernel<<<kB * kC, 256, 0, stream>>>(Y, stats);
        norm_relu_kernel<<<elem4 / 256, 256, 0, stream>>>(Y, stats, T);
        conv_kernel<<<convGrid, THREADS, 0, stream>>>(T, gemm, Wt2, b2, Y);
        stats_kernel<<<kB * kC, 256, 0, stream>>>(Y, stats);
        norm_add_relu_kernel<<<elem4 / 256, 256, 0, stream>>>(Y, stats, P, out);
        P = out;                                 // blocks 2,3 run in place on d_out
    }
}
